// Attention_60464549593105
// MI455X (gfx1250) — matrix operand order repaired and checked by reading, not compiled, not run on hardware
//
#include <hip/hip_runtime.h>
#include <math.h>

// Shapes: B=4, D=128(K-dim of proj), E2=128, S=64, H*W=4096
#define HW_ 4096
#define SHW_ (64 * 4096)           // elements between consecutive e planes
#define ESTRIDE_ ((size_t)SHW_)
#define BSTRIDE_ ((size_t)128 * SHW_)

typedef __attribute__((ext_vector_type(16))) __bf16 bf16x16;
typedef __attribute__((ext_vector_type(2)))  __bf16 bf16x2;
typedef __attribute__((ext_vector_type(8)))  float  floatx8;

#if __has_builtin(__builtin_amdgcn_tanhf)
#define TANHF(x) __builtin_amdgcn_tanhf(x)
#else
#define TANHF(x) tanhf(x)
#endif

#if __has_builtin(__builtin_amdgcn_exp2f)
#define EXP2F(x) __builtin_amdgcn_exp2f(x)
#else
#define EXP2F(x) exp2f(x)
#endif

__device__ __forceinline__ unsigned short f2bf(float f) {
  union { float f; unsigned int u; } c; c.f = f;
  unsigned int u = c.u;
  u += 0x7fffu + ((u >> 16) & 1u);   // round-to-nearest-even
  return (unsigned short)(u >> 16);
}

// Pack two f32 -> bf16 pair.
//  1) hw packed cvt if the builtin exists (not on this toolchain per probe)
//  2) v_perm_b32 path: 2x v_add + 1x v_perm (round-half-up)
//  3) manual RNE fallback
__device__ __forceinline__ unsigned int pkbf(float lo, float hi) {
#if __has_builtin(__builtin_amdgcn_cvt_pk_bf16_f32)
  bf16x2 v = __builtin_amdgcn_cvt_pk_bf16_f32(lo, hi);
  union { bf16x2 v; unsigned int u; } c; c.v = v;
  return c.u;
#elif __has_builtin(__builtin_amdgcn_perm)
  union { float f; unsigned int u; } a, b;
  a.f = lo; b.f = hi;
  // result bytes {hi[3],hi[2],lo[3],lo[2]}: S0=hi -> sel 7,6; S1=lo -> sel 3,2
  return __builtin_amdgcn_perm(b.u + 0x8000u, a.u + 0x8000u, 0x07060302u);
#else
  return (unsigned int)f2bf(lo) | ((unsigned int)f2bf(hi) << 16);
#endif
}

union V16 {
  bf16x16 bf;
  unsigned short us[16];
  unsigned int d[8];
  uint4 q4[2];
};

// ---------------------------------------------------------------------------
// Kernel 1: P[b,k,hw] = sum_d hidden[b,d,hw] * Wh[k,d] + b_att[k]
// One block of 128 threads handles 128 hw positions for one b.
// ---------------------------------------------------------------------------
__global__ void __launch_bounds__(128) proj_h_kernel(
    const float* __restrict__ hidden, const float* __restrict__ W_att,
    const float* __restrict__ b_att, float* __restrict__ P) {
  __shared__ float h[128][129];
  const int t = threadIdx.x;
  const int b = blockIdx.x >> 5;            // 32 blocks per batch
  const int hw0 = (blockIdx.x & 31) << 7;   // 128 positions per block
  for (int d = 0; d < 128; ++d)
    h[d][t] = hidden[((size_t)(b * 128 + d) << 12) + hw0 + t];
  __syncthreads();
  for (int k = 0; k < 128; ++k) {
    const float* wr = W_att + k * 256;      // Wh = W_att[:, :128], row stride 256
    float acc = b_att[k];
#pragma unroll 8
    for (int d = 0; d < 128; ++d) acc = fmaf(h[d][t], wr[d], acc);
    P[((size_t)(b * 128 + k) << 12) + hw0 + t] = acc;
  }
}

// ---------------------------------------------------------------------------
// Kernel 2: fused  proj_e (bf16 WMMA, f32 acc) + tanh + W_v dot + softmax(S)
// One WG (256 thr = 8 waves) per (b, 16-wide hw tile); wave w owns s=8w..8w+7.
// ---------------------------------------------------------------------------
__global__ void __launch_bounds__(256) att_kernel(
    const float* __restrict__ enc, const float* __restrict__ W_att,
    const float* __restrict__ Wv, const float* __restrict__ P,
    float* __restrict__ out) {
  // We bf16: rows padded to 136 ushorts (272B = 17*16) -> 16B-aligned halves,
  // bank stride 68 dwords == 4 mod 64 (conflict-free wide loads).
  __shared__ __align__(16) unsigned short we[128 * 136];
  // P tile transposed to n-major, stride 132 (== 4 mod 64): per-lane float4s.
  __shared__ __align__(16) float pl[16 * 132];
  __shared__ __align__(16) float wv[128];
  __shared__ float sc[64 * 16];             // scores[s][n]
  __shared__ float red[32];                 // per-column max / inv-sum

  const int t   = threadIdx.x;
  const int b   = blockIdx.x >> 8;
  const int hw0 = (blockIdx.x & 255) << 4;

  // Stage We = W_att[:,128:] as bf16 (paired convert, dword stores).
  for (int p = t; p < 64 * 128; p += 256) {
    const int k = p >> 6, e2 = (p & 63) << 1;
    const float f0 = W_att[k * 256 + 128 + e2];
    const float f1 = W_att[k * 256 + 128 + e2 + 1];
    *(unsigned int*)&we[k * 136 + e2] = pkbf(f0, f1);
  }
  // Stage P (proj_h + bias), transposed: pl[n][k].
  for (int idx = t; idx < 128 * 16; idx += 256) {
    const int k = idx >> 4, n = idx & 15;
    pl[n * 132 + k] = P[((size_t)(b * 128 + k) << 12) + hw0 + n];
  }
  if (t < 128) wv[t] = Wv[t];
  __syncthreads();

  const int lane = t & 31;
  const int wave = t >> 5;
  const int n    = lane & 15;   // matrix column / hw offset
  const int hi   = lane >> 4;   // half-wave select
  const size_t encb = (size_t)b * BSTRIDE_ + hw0 + n;

  for (int si = 0; si < 8; ++si) {
    const int s = wave * 8 + si;
    // Load B tiles (32e x 16hw) straight into WMMA layout; convert in pairs
    // (operand elements 2j,2j+1 are adjacent e planes).
    V16 Bt[4];
#pragma unroll
    for (int kt = 0; kt < 4; ++kt) {
      const int ebase = kt * 32 + hi * 16;
      const size_t base = encb + (size_t)s * HW_;
#pragma unroll
      for (int j = 0; j < 8; ++j) {
        const float f0 = enc[base + (size_t)(ebase + 2 * j)     * ESTRIDE_];
        const float f1 = enc[base + (size_t)(ebase + 2 * j + 1) * ESTRIDE_];
        Bt[kt].d[j] = pkbf(f0, f1);
      }
    }
    float part = 0.0f;
#pragma unroll
    for (int mt = 0; mt < 8; ++mt) {
      floatx8 acc = {};
#pragma unroll
      for (int kt = 0; kt < 4; ++kt) {
        // A tile 16k x 32e: per-lane two 16B LDS loads (e_off 16*hi + 0..15, contiguous).
        V16 At;
        const unsigned short* rp = &we[(mt * 16 + n) * 136 + kt * 32 + hi * 16];
        At.q4[0] = *(const uint4*)(rp);
        At.q4[1] = *(const uint4*)(rp + 8);
        acc = __builtin_amdgcn_wmma_f32_16x16x32_bf16(
            false, At.bf, false, Bt[kt].bf, (short)0, acc, false, false);
      }
      // Consume C tile: lane holds rows kb..kb+7 of column n.
      const int kb = mt * 16 + hi * 8;
      const float4 p0 = *(const float4*)&pl[n * 132 + kb];
      const float4 p1 = *(const float4*)&pl[n * 132 + kb + 4];
      const float4 w0 = *(const float4*)&wv[kb];
      const float4 w1 = *(const float4*)&wv[kb + 4];
      part = fmaf(w0.x, TANHF(acc[0] + p0.x), part);
      part = fmaf(w0.y, TANHF(acc[1] + p0.y), part);
      part = fmaf(w0.z, TANHF(acc[2] + p0.z), part);
      part = fmaf(w0.w, TANHF(acc[3] + p0.w), part);
      part = fmaf(w1.x, TANHF(acc[4] + p1.x), part);
      part = fmaf(w1.y, TANHF(acc[5] + p1.y), part);
      part = fmaf(w1.z, TANHF(acc[6] + p1.z), part);
      part = fmaf(w1.w, TANHF(acc[7] + p1.w), part);
    }
    part += __shfl_xor(part, 16, 32);  // fold the two k-halves
    if (hi == 0) sc[s * 16 + n] = part;
  }
  __syncthreads();

  // Softmax over s for each of the 16 hw columns.
  if (t < 16) {
    float mx = -3.0e38f;
    for (int s = 0; s < 64; ++s) mx = fmaxf(mx, sc[s * 16 + t]);
    float sum = 0.0f;
    for (int s = 0; s < 64; ++s)
      sum += EXP2F((sc[s * 16 + t] - mx) * 1.4426950408889634f);
    red[t]      = mx;
    red[16 + t] = 1.0f / sum;
  }
  __syncthreads();

  const size_t outb = (size_t)b * (64 * (size_t)HW_) + hw0;
  for (int idx = t; idx < 64 * 16; idx += 256) {
    const int s = idx >> 4, nn = idx & 15;
    out[outb + (size_t)s * HW_ + nn] =
        EXP2F((sc[idx] - red[nn]) * 1.4426950408889634f) * red[16 + nn];
  }
}

extern "C" void kernel_launch(void* const* d_in, const int* in_sizes, int n_in,
                              void* d_out, int out_size, void* d_ws, size_t ws_size,
                              hipStream_t stream) {
  const float* hidden = (const float*)d_in[0];   // (4,128,64,64)
  const float* enc    = (const float*)d_in[1];   // (4,128,64,64,64)
  const float* W_att  = (const float*)d_in[2];   // (128,256)
  const float* b_att  = (const float*)d_in[3];   // (128,)
  const float* W_v    = (const float*)d_in[4];   // (1,128)
  float* out = (float*)d_out;                    // (4,64,64,64)
  float* P   = (float*)d_ws;                     // 4*128*4096 floats = 8 MB

  proj_h_kernel<<<128, 128, 0, stream>>>(hidden, W_att, b_att, P);
  att_kernel<<<1024, 256, 0, stream>>>(enc, W_att, W_v, P, out);
}
